// MultiHeadAttention_38860864094296
// MI455X (gfx1250) — matrix operand order repaired and checked by reading, not compiled, not run on hardware
//
#include <hip/hip_runtime.h>
#include <hip/hip_bf16.h>

#define S_LEN 2048
#define HIDDIM 2048
#define NHEAD 16
#define HDIM 128
#define QKV_N (3 * HIDDIM)   // 6144

typedef __attribute__((ext_vector_type(16))) __bf16 v16bf;
typedef __attribute__((ext_vector_type(8)))  float  v8f;
typedef __attribute__((ext_vector_type(4)))  int    v4i;
typedef unsigned int  uint32;
typedef unsigned short u16;

#if __has_builtin(__builtin_amdgcn_global_load_async_to_lds_b128) && \
    __has_builtin(__builtin_amdgcn_s_wait_asynccnt)
#define HAVE_ASYNC_LDS 1
#endif

// ---------- scalar helpers ----------
__device__ __forceinline__ u16 f32_to_bf16_rne(float f) {
    union { float f; uint32 u; } x; x.f = f;
    uint32 r = x.u + 0x7fffu + ((x.u >> 16) & 1u);
    return (u16)(r >> 16);
}
__device__ __forceinline__ __bf16 as_bf16(u16 h) {
    return __builtin_bit_cast(__bf16, h);
}

#ifdef HAVE_ASYNC_LDS
__device__ __forceinline__ void async_cp16(const void* g, void* l) {
    __builtin_amdgcn_global_load_async_to_lds_b128(
        (__attribute__((address_space(1))) v4i*)g,
        (__attribute__((address_space(3))) v4i*)l,
        0, 0);
}
#endif

// ---------- WMMA fragment loaders (wave32; contiguous k = 16*(lane/16) + s in A and B, not the ISA table order) ----------
// A-matrix 16x32 bf16: lane m = lane&15; lanes 0-15 hold K {0..15},
// lanes 16-31 hold K {16..31}.
__device__ __forceinline__ v16bf load_a_frag(const u16* tile, int ldk, int lane) {
    const int m  = lane & 15;
    const int kb = lane & 16;             // 0 or 16
    const u16* row = tile + (size_t)m * ldk;
    v16bf a;
#pragma unroll
    for (int i = 0; i < 8; ++i) a[i]     = as_bf16(row[kb + i]);
#pragma unroll
    for (int i = 0; i < 8; ++i) a[8 + i] = as_bf16(row[kb + 8 + i]);
    return a;
}

// B-matrix 32x16 bf16 where B[k][n] = tile[n][k] (tile rows are N, stride ldk).
__device__ __forceinline__ v16bf load_b_frag(const u16* tile, int ldk, int lane) {
    const int n  = lane & 15;
    const int kb = lane & 16;             // 0 or 16
    const u16* row = tile + (size_t)n * ldk;
    v16bf b;
#pragma unroll
    for (int i = 0; i < 16; ++i) b[i] = as_bf16(row[kb + i]);
    return b;
}

// B-matrix 32x16 bf16 where B[k][n] = tile[k][col0 + n] (tile rows are K, stride ldn).
__device__ __forceinline__ v16bf load_bk_frag(const u16* tile, int ldn, int col0, int lane) {
    const int n  = col0 + (lane & 15);
    const int kb = lane & 16;
    v16bf b;
#pragma unroll
    for (int i = 0; i < 16; ++i) b[i] = as_bf16(tile[(size_t)(kb + i) * ldn + n]);
    return b;
}

// ---------- f32 -> bf16 weight conversion ----------
__global__ void cvt_bf16_kernel(const float* __restrict__ in, u16* __restrict__ out, int n) {
    int i = blockIdx.x * blockDim.x + threadIdx.x;
    const int stride = gridDim.x * blockDim.x;
    for (; i < n; i += stride) out[i] = f32_to_bf16_rne(in[i]);
}

// ---------- LayerNorm over HID, output bf16 ----------
__global__ __launch_bounds__(256) void ln_kernel(const float* __restrict__ x,
                                                 const float* __restrict__ w,
                                                 const float* __restrict__ b,
                                                 u16* __restrict__ out) {
    __shared__ float red[256];
    const int t = threadIdx.x;
    const size_t base = (size_t)blockIdx.x * HIDDIM;

    float v[8]; float s = 0.f;
#pragma unroll
    for (int i = 0; i < 8; ++i) { v[i] = x[base + t + i * 256]; s += v[i]; }
    red[t] = s; __syncthreads();
    for (int off = 128; off > 0; off >>= 1) { if (t < off) red[t] += red[t + off]; __syncthreads(); }
    const float mu = red[0] * (1.f / HIDDIM);
    __syncthreads();

    float sq = 0.f;
#pragma unroll
    for (int i = 0; i < 8; ++i) { float d = v[i] - mu; sq += d * d; }
    red[t] = sq; __syncthreads();
    for (int off = 128; off > 0; off >>= 1) { if (t < off) red[t] += red[t + off]; __syncthreads(); }
    const float rstd = rsqrtf(red[0] * (1.f / HIDDIM) + 1e-5f);

#pragma unroll
    for (int i = 0; i < 8; ++i) {
        const int c = t + i * 256;
        out[base + c] = f32_to_bf16_rne((v[i] - mu) * rstd * w[c] + b[c]);
    }
}

// ---------- bf16 WMMA GEMM: C[M,N] = A[M,K] * W[N,K]^T + bias ----------
// Block: 256 threads (8 waves, 4x2 grid). Block tile 128x128, wave tile 32x64.
template <bool OUT_BF16>
__global__ __launch_bounds__(256) void gemm_bf16_kernel(const u16* __restrict__ A,
                                                        const u16* __restrict__ W,
                                                        const float* __restrict__ bias,
                                                        void* __restrict__ C,
                                                        int M, int N, int K) {
    __shared__ u16 lA[128 * 32];
    __shared__ u16 lB[128 * 32];
    const int tid  = threadIdx.x;
    const int lane = tid & 31;
    const int wave = tid >> 5;
    const int wr   = wave & 3;        // M strip of 32 rows
    const int wc   = wave >> 2;       // N half of 64 cols
    const int n0   = blockIdx.x * 128;
    const int m0   = blockIdx.y * 128;

    v8f acc[2][4];
#pragma unroll
    for (int s = 0; s < 2; ++s)
#pragma unroll
        for (int i = 0; i < 4; ++i)
#pragma unroll
            for (int j = 0; j < 8; ++j) acc[s][i][j] = 0.f;

    for (int kk = 0; kk < K; kk += 32) {
        // stage 128x32 A-tile and 128x32 W-tile into LDS (16B chunks)
#ifdef HAVE_ASYNC_LDS
#pragma unroll
        for (int i = 0; i < 2; ++i) {
            const int idx = tid + i * 256;             // 512 chunks per tile
            const int r = idx >> 2, c = (idx & 3) * 8; // 8 elems per chunk
            async_cp16(A + (size_t)(m0 + r) * K + kk + c, lA + r * 32 + c);
            async_cp16(W + (size_t)(n0 + r) * K + kk + c, lB + r * 32 + c);
        }
        __builtin_amdgcn_s_wait_asynccnt(0);
#else
#pragma unroll
        for (int i = 0; i < 2; ++i) {
            const int idx = tid + i * 256;
            const int r = idx >> 2, c = idx & 3;
            ((uint4*)lA)[idx] = ((const uint4*)(A + (size_t)(m0 + r) * K + kk))[c];
            ((uint4*)lB)[idx] = ((const uint4*)(W + (size_t)(n0 + r) * K + kk))[c];
        }
#endif
        __syncthreads();

        const u16* lAw = lA + (wr * 32) * 32;
        const v16bf a0 = load_a_frag(lAw,           32, lane);
        const v16bf a1 = load_a_frag(lAw + 16 * 32, 32, lane);
#pragma unroll
        for (int nc = 0; nc < 4; ++nc) {
            const v16bf b = load_b_frag(lB + (wc * 64 + nc * 16) * 32, 32, lane);
            acc[0][nc] = __builtin_amdgcn_wmma_f32_16x16x32_bf16(
                false, a0, false, b, (short)0, acc[0][nc], false, false);
            acc[1][nc] = __builtin_amdgcn_wmma_f32_16x16x32_bf16(
                false, a1, false, b, (short)0, acc[1][nc], false, false);
        }
        __syncthreads();
    }

    const int hi8 = (lane >> 4) << 3;
    const int nl  = lane & 15;
#pragma unroll
    for (int sub = 0; sub < 2; ++sub) {
        const int rbase = m0 + wr * 32 + sub * 16 + hi8;
#pragma unroll
        for (int nc = 0; nc < 4; ++nc) {
            const int col = n0 + wc * 64 + nc * 16 + nl;
            const float bv = bias[col];
#pragma unroll
            for (int r = 0; r < 8; ++r) {
                const float v = acc[sub][nc][r] + bv;
                if (OUT_BF16) ((u16*)C)[(size_t)(rbase + r) * N + col] = f32_to_bf16_rne(v);
                else          ((float*)C)[(size_t)(rbase + r) * N + col] = v;
            }
        }
    }
}

// ---------- flash-style causal attention ----------
// Grid: (S/64, NHEAD). Block: 128 threads = 4 waves; wave w owns query rows
// qb + 16w .. +15 of head blockIdx.y. All waves share staged K/V tiles (32 keys).
__global__ __launch_bounds__(128) void attn_kernel(const u16* __restrict__ mixed,
                                                   u16* __restrict__ ctx) {
    __shared__ u16 lK[32 * 128];
    __shared__ u16 lV[32 * 128];
    __shared__ u16 lP[4 * 16 * 32];

    const int tid  = threadIdx.x;
    const int lane = tid & 31;
    const int wave = tid >> 5;
    const int h    = blockIdx.y;
    const int qb   = blockIdx.x * 64;
    const int q0   = qb + wave * 16;
    const int qcol = h * HDIM;
    const int kcol = HIDDIM + h * HDIM;
    const int vcol = 2 * HIDDIM + h * HDIM;
    const float scale = 0.08838834764831845f;   // 1/sqrt(128)

    // Q fragments stay in registers (16 x 128 -> 4 chunks of K=32)
    v16bf qf[4];
#pragma unroll
    for (int c = 0; c < 4; ++c)
        qf[c] = load_a_frag(mixed + (size_t)q0 * QKV_N + qcol + c * 32, QKV_N, lane);

    v8f o[8];
    float mrow[8], lrow[8];
#pragma unroll
    for (int nc = 0; nc < 8; ++nc)
#pragma unroll
        for (int r = 0; r < 8; ++r) o[nc][r] = 0.f;
#pragma unroll
    for (int r = 0; r < 8; ++r) { mrow[r] = -1e30f; lrow[r] = 0.f; }

    const int hi8 = (lane >> 4) << 3;
    const int nl  = lane & 15;
    u16* lPw = lP + wave * 16 * 32;

    const int ntiles = (qb + 64) >> 5;           // causal key-tile limit for block
    for (int j = 0; j < ntiles; ++j) {
        const int k0 = j * 32;
        // stage 32x128 K and V tiles (16B chunks; 512 chunks each)
#ifdef HAVE_ASYNC_LDS
#pragma unroll
        for (int i = 0; i < 4; ++i) {
            const int idx = tid + i * 128;
            const int r = idx >> 4, c = (idx & 15) * 8;
            async_cp16(mixed + (size_t)(k0 + r) * QKV_N + kcol + c, lK + r * 128 + c);
            async_cp16(mixed + (size_t)(k0 + r) * QKV_N + vcol + c, lV + r * 128 + c);
        }
        __builtin_amdgcn_s_wait_asynccnt(0);
#else
#pragma unroll
        for (int i = 0; i < 4; ++i) {
            const int idx = tid + i * 128;
            const int r = idx >> 4, c = idx & 15;
            ((uint4*)lK)[idx] = ((const uint4*)(mixed + (size_t)(k0 + r) * QKV_N + kcol))[c];
            ((uint4*)lV)[idx] = ((const uint4*)(mixed + (size_t)(k0 + r) * QKV_N + vcol))[c];
        }
#endif
        __syncthreads();

        if (k0 <= q0 + 15) {                      // wave-uniform predicate (EXEC stays full)
            // scores: Q(16x128) . K^T -> two 16x16 f32 tiles
            v8f s[2];
#pragma unroll
            for (int sc = 0; sc < 2; ++sc) {
#pragma unroll
                for (int r = 0; r < 8; ++r) s[sc][r] = 0.f;
#pragma unroll
                for (int c = 0; c < 4; ++c) {
                    const v16bf b = load_b_frag(lK + sc * 16 * 128 + c * 32, 128, lane);
                    s[sc] = __builtin_amdgcn_wmma_f32_16x16x32_bf16(
                        false, qf[c], false, b, (short)0, s[sc], false, false);
                }
            }
            // scale + causal mask
#pragma unroll
            for (int sc = 0; sc < 2; ++sc)
#pragma unroll
                for (int r = 0; r < 8; ++r) {
                    const int qrow = q0 + r + hi8;
                    const int krow = k0 + sc * 16 + nl;
                    const float v = s[sc][r] * scale;
                    s[sc][r] = (krow <= qrow) ? v : -1e30f;
                }
            // online softmax (row reductions across 16-lane halves)
#pragma unroll
            for (int r = 0; r < 8; ++r) {
                float tmax = fmaxf(s[0][r], s[1][r]);
#pragma unroll
                for (int msk = 1; msk < 16; msk <<= 1)
                    tmax = fmaxf(tmax, __shfl_xor(tmax, msk, 32));
                const float mnew = fmaxf(mrow[r], tmax);
                const float p0 = __expf(s[0][r] - mnew);
                const float p1 = __expf(s[1][r] - mnew);
                float psum = p0 + p1;
#pragma unroll
                for (int msk = 1; msk < 16; msk <<= 1)
                    psum += __shfl_xor(psum, msk, 32);
                const float alpha = __expf(mrow[r] - mnew);
                lrow[r] = lrow[r] * alpha + psum;
                mrow[r] = mnew;
#pragma unroll
                for (int nc = 0; nc < 8; ++nc) o[nc][r] *= alpha;
                // stash P as bf16 in wave-private LDS (C-layout -> A-layout transpose)
                lPw[(r + hi8) * 32 + nl]      = f32_to_bf16_rne(p0);
                lPw[(r + hi8) * 32 + 16 + nl] = f32_to_bf16_rne(p1);
            }
            // P(16x32) . V(32x128): same-wave LDS RAW is in-order
            const v16bf pa = load_a_frag(lPw, 32, lane);
#pragma unroll
            for (int nc = 0; nc < 8; ++nc) {
                const v16bf bv = load_bk_frag(lV, 128, nc * 16, lane);
                o[nc] = __builtin_amdgcn_wmma_f32_16x16x32_bf16(
                    false, pa, false, bv, (short)0, o[nc], false, false);
            }
        }
        __syncthreads();
    }

    // epilogue: normalize, write bf16 ctx [S, HID]
#pragma unroll
    for (int r = 0; r < 8; ++r) {
        const float inv = 1.f / lrow[r];
        const int row = q0 + r + hi8;
#pragma unroll
        for (int nc = 0; nc < 8; ++nc)
            ctx[(size_t)row * HIDDIM + h * HDIM + nc * 16 + nl] =
                f32_to_bf16_rne(o[nc][r] * inv);
    }
}

// ---------- host launcher ----------
extern "C" void kernel_launch(void* const* d_in, const int* in_sizes, int n_in,
                              void* d_out, int out_size, void* d_ws, size_t ws_size,
                              hipStream_t stream) {
    (void)in_sizes; (void)n_in; (void)out_size; (void)ws_size;
    const float* hidden  = (const float*)d_in[0];
    const float* lnw     = (const float*)d_in[1];
    const float* lnb     = (const float*)d_in[2];
    const float* qkvw_f  = (const float*)d_in[3];
    const float* qkvb    = (const float*)d_in[4];
    const float* projw_f = (const float*)d_in[5];
    const float* projb   = (const float*)d_in[6];
    float* out = (float*)d_out;

    u16* ws = (u16*)d_ws;
    size_t off = 0;
    u16* ln_x  = ws + off; off += (size_t)S_LEN * HIDDIM;   // 2048x2048 bf16
    u16* qkvw  = ws + off; off += (size_t)QKV_N * HIDDIM;   // 6144x2048 bf16
    u16* projw = ws + off; off += (size_t)HIDDIM * HIDDIM;  // 2048x2048 bf16
    u16* mixed = ws + off; off += (size_t)S_LEN * QKV_N;    // 2048x6144 bf16
    u16* ctx   = ws + off;                                  // 2048x2048 bf16

    cvt_bf16_kernel<<<2048, 256, 0, stream>>>(qkvw_f, qkvw, QKV_N * HIDDIM);
    cvt_bf16_kernel<<<2048, 256, 0, stream>>>(projw_f, projw, HIDDIM * HIDDIM);
    ln_kernel<<<S_LEN, 256, 0, stream>>>(hidden, lnw, lnb, ln_x);

    gemm_bf16_kernel<true><<<dim3(QKV_N / 128, S_LEN / 128), 256, 0, stream>>>(
        ln_x, qkvw, qkvb, mixed, S_LEN, QKV_N, HIDDIM);

    attn_kernel<<<dim3(S_LEN / 64, NHEAD), 128, 0, stream>>>(mixed, ctx);

    gemm_bf16_kernel<false><<<dim3(HIDDIM / 128, S_LEN / 128), 256, 0, stream>>>(
        ctx, projw, projb, out, S_LEN, HIDDIM, HIDDIM);
}
